// LRU_80161269613382
// MI455X (gfx1250) — compile-verified, matrix operand order checked by reading, not run on hardware
//
#include <hip/hip_runtime.h>
#include <math.h>

// ---------------------------------------------------------------------------
// LRU layer for MI455X (gfx1250, wave32, WMMA bf16, async global->LDS staging)
//   IN=256, OUT=256, HID=512, BATCH=16, T=4096  ->  M = 65536 rows
//
// Pipeline:
//   W1[2h+p,k] = gamma[h]*{B_re,B_im}[h,k] (bf16); lam, lam^SEGLEN
//   drive      = Xbf16 * W1^T + bias1                 [65536 x 1024] f32
//   3-phase blocked scan (32 segments x 128 steps)  -> Z (bf16, interleaved)
//   W2[o,j]    = [interleaved C_re|-C_im | D]         bf16
//   Y          = Z * W2^T + bias_out                  [65536 x 256]  f32
// ---------------------------------------------------------------------------

#define IN_DIM   256
#define OUT_DIM  256
#define HID_DIM  512
#define BATCH    16
#define TSTEPS   4096
#define MROWS    (BATCH * TSTEPS)          // 65536
#define K1       IN_DIM                    // 256
#define N1       (2 * HID_DIM)             // 1024 (re/im interleaved)
#define K2       (2 * HID_DIM + IN_DIM)    // 1280 (H interleaved | X)
#define N2       OUT_DIM                   // 256
#define NCHAN    (BATCH * HID_DIM)         // 8192
#define SEGS     32
#define SEGLEN   (TSTEPS / SEGS)           // 128

typedef __attribute__((ext_vector_type(16))) __bf16 v16bf;
typedef __attribute__((ext_vector_type(8)))  __bf16 v8bf;
typedef __attribute__((ext_vector_type(2)))  __bf16 v2bf;
typedef __attribute__((ext_vector_type(8)))  float  v8f;

// GEMM tiling (wave32): block = 256 thr = 8 waves; block tile 64(M) x 256(N);
// wave grid 2x4 -> each wave owns 32x64 = 2x4 WMMA tiles; KC=64 -> 16 WMMA
// per barrier pair. Double-buffered LDS: 2*(64+256)*72*2B = 90 KB (< 320 KB/WGP).
#define BM   64
#define BN   256
#define KC   64
#define LDT  (KC + 8)    // 144B row stride -> 16 rows hit distinct bank groups

// ---------------------------------------------------------------------------
// gfx1250 async global->LDS copy (16B per lane). Tracked by ASYNCcnt.
// LDS dest VGPR = low 32 bits of the flat shared address (ISA 10.2).
// ---------------------------------------------------------------------------
__device__ __forceinline__ void lru_async_copy16(const __bf16* g, __bf16* l) {
    unsigned lds_off = (unsigned)(unsigned long long)(uintptr_t)l;
    unsigned long long ga = (unsigned long long)(uintptr_t)g;
    asm volatile("global_load_async_to_lds_b128 %0, %1, off"
                 :: "v"(lds_off), "v"(ga) : "memory");
}
#define LRU_ASYNC_WAIT(n) asm volatile("s_wait_asynccnt " #n ::: "memory")

// ---------------------------------------------------------------------------
// Fragment load from LDS per the CDNA5 16-bit A-matrix 16x32 layout:
//  lanes 0-15 : row M = lane,     K in {k0..k0+7}   U {k0+16..k0+23}
//  lanes 16-31: row M = lane-16,  K in {k0+8..k0+15} U {k0+24..k0+31}
// B (row-major W[N,K]) uses the symmetric pattern with N = lane&15.
// ---------------------------------------------------------------------------
__device__ __forceinline__ v16bf lru_load_frag(const __bf16* base, int k0, int lane) {
    const int m   = lane & 15;
    const int sel = lane >> 4;
    const __bf16* p = base + m * LDT + k0 + sel * 8;
    v8bf lo = *(const v8bf*)(p);               // ds_load_b128
    v8bf hi = *(const v8bf*)(p + 16);
    return __builtin_shufflevector(lo, hi, 0,1,2,3,4,5,6,7,8,9,10,11,12,13,14,15);
}

// ---------------------------------------------------------------------------
// Stage one 64xKC A chunk + 256xKC B chunk into LDS with async copies.
// Each thread issues exactly 10 async ops (2 for A, 8 for B).
// ---------------------------------------------------------------------------
__device__ __forceinline__ void lru_stage(const __bf16* __restrict__ A, int lda,
                                          const __bf16* __restrict__ W, int K,
                                          int row0, int col0, int kc,
                                          __bf16* lA, __bf16* lB, int tid)
{
    #pragma unroll
    for (int i = 0; i < (BM * (KC / 8)) / 256; ++i) {   // A: 64 x 8 = 512 txns
        const int idx = tid + i * 256;
        const int r   = idx >> 3;
        const int c8  = (idx & 7) * 8;
        lru_async_copy16(&A[(size_t)(row0 + r) * lda + kc + c8], &lA[r * LDT + c8]);
    }
    #pragma unroll
    for (int i = 0; i < (BN * (KC / 8)) / 256; ++i) {   // B: 256 x 8 = 2048 txns
        const int idx = tid + i * 256;
        const int r   = idx >> 3;
        const int c8  = (idx & 7) * 8;
        lru_async_copy16(&W[(size_t)(col0 + r) * K + kc + c8], &lB[r * LDT + c8]);
    }
}

// ---------------------------------------------------------------------------
// Generic bf16 GEMM: out[M,N](f32) = A[M,K](bf16, row stride lda) * W[N,K]^T + bias
// Double-buffered async LDS staging; 16 WMMA per K chunk.
// ---------------------------------------------------------------------------
__global__ __launch_bounds__(256)
void lru_gemm_bf16(const __bf16* __restrict__ A, int lda,
                   const __bf16* __restrict__ W,
                   const float*  __restrict__ bias,
                   float* __restrict__ out,
                   int M, int N, int K)
{
    __shared__ __align__(16) __bf16 lA[2][BM * LDT];
    __shared__ __align__(16) __bf16 lB[2][BN * LDT];

    const int tid  = threadIdx.x;
    const int wave = tid >> 5;
    const int lane = tid & 31;
    const int wm   = wave >> 2;     // 0..1 -> M sub-block of 32
    const int wn   = wave & 3;      // 0..3 -> N sub-block of 64

    const int row0 = blockIdx.y * BM;
    const int col0 = blockIdx.x * BN;

    v8f acc[2][4] = {};

    int buf = 0;
    lru_stage(A, lda, W, K, row0, col0, 0, lA[0], lB[0], tid);

    for (int kc = 0; kc < K; kc += KC) {
        const bool has_next = (kc + KC) < K;
        if (has_next) {
            if (kc + 2 * KC < K) {   // prefetch 2 chunks ahead
                __builtin_prefetch(A + (size_t)row0 * lda + kc + 2 * KC, 0, 3);
                __builtin_prefetch(W + (size_t)col0 * K + kc + 2 * KC, 0, 3);
            }
            lru_stage(A, lda, W, K, row0, col0, kc + KC, lA[buf ^ 1], lB[buf ^ 1], tid);
            LRU_ASYNC_WAIT(10);  // drain the older 10 (current buf); next stays in flight
        } else {
            LRU_ASYNC_WAIT(0);
        }
        __syncthreads();

        #pragma unroll
        for (int k0 = 0; k0 < KC; k0 += 32) {
            v16bf afrag[2], bfrag[4];
            #pragma unroll
            for (int t = 0; t < 2; ++t)
                afrag[t] = lru_load_frag(&lA[buf][(wm * 32 + t * 16) * LDT], k0, lane);
            #pragma unroll
            for (int t = 0; t < 4; ++t)
                bfrag[t] = lru_load_frag(&lB[buf][(wn * 64 + t * 16) * LDT], k0, lane);
            #pragma unroll
            for (int tm = 0; tm < 2; ++tm)
                #pragma unroll
                for (int tn = 0; tn < 4; ++tn)
                    acc[tm][tn] = __builtin_amdgcn_wmma_f32_16x16x32_bf16(
                        false, afrag[tm], false, bfrag[tn],
                        (short)0, acc[tm][tn], false, false);
        }

        __syncthreads();         // all waves done reading buf before overwrite
        buf ^= 1;
    }

    // epilogue: C/D layout -> lane holds N = lane&15, M = vgpr + 8*(lane>=16)
    const int nlane = lane & 15;
    const int moff  = (lane >> 4) * 8;
    const int mbase = row0 + wm * 32;
    const int nbase = col0 + wn * 64;
    #pragma unroll
    for (int tm = 0; tm < 2; ++tm)
        #pragma unroll
        for (int tn = 0; tn < 4; ++tn) {
            const int col = nbase + tn * 16 + nlane;
            const float b = bias[col];
            #pragma unroll
            for (int r = 0; r < 8; ++r) {
                const int row = mbase + tm * 16 + moff + r;
                out[(size_t)row * N + col] = acc[tm][tn][r] + b;
            }
        }
}

// ---------------------------------------------------------------------------
// Weight pack 1 (interleaved): W1[2h+p,k] = gamma[h]*{B_re,B_im}[h,k] (bf16)
// bias1[2h+p]; lam = lambda (interleaved); lamL = lambda^SEGLEN (interleaved)
// ---------------------------------------------------------------------------
__global__ void lru_pack_w1(const float* __restrict__ nu_log,
                            const float* __restrict__ theta_log,
                            const float* __restrict__ B_re,
                            const float* __restrict__ B_im,
                            const float* __restrict__ bh_re,
                            const float* __restrict__ bh_im,
                            __bf16* __restrict__ W1,
                            float* __restrict__ bias1,
                            float* __restrict__ lam,
                            float* __restrict__ lamL)
{
    const int n = blockIdx.x * blockDim.x + threadIdx.x;
    if (n >= N1) return;
    const int h  = n >> 1;
    const int im = n & 1;
    const float nu  = __expf(nu_log[h]);
    const float mag = __expf(-nu);
    const float gam = sqrtf(fmaxf(0.f, 1.f - mag * mag));
    const float* src = im ? B_im : B_re;
    for (int k = 0; k < K1; ++k)
        W1[(size_t)n * K1 + k] = (__bf16)(gam * src[h * K1 + k]);
    bias1[n] = im ? bh_im[h] : bh_re[h];
    if (!im) {
        const float th = __expf(theta_log[h]);
        lam[2 * h]     = mag * __cosf(th);
        lam[2 * h + 1] = mag * __sinf(th);
        const float magL = __expf(-(float)SEGLEN * nu);   // mag^SEGLEN, exact
        const float thL  = (float)SEGLEN * th;
        lamL[2 * h]     = magL * __cosf(thL);
        lamL[2 * h + 1] = magL * __sinf(thL);
    }
}

// ---------------------------------------------------------------------------
// Weight pack 2: W2[o,2h]=C_re[o,h], W2[o,2h+1]=-C_im[o,h], W2[o,1024+i]=D[o,i]
// ---------------------------------------------------------------------------
__global__ void lru_pack_w2(const float* __restrict__ C_re,
                            const float* __restrict__ C_im,
                            const float* __restrict__ D,
                            __bf16* __restrict__ W2)
{
    const int id = blockIdx.x * blockDim.x + threadIdx.x;
    if (id >= N2 * K2) return;
    const int o = id / K2;
    const int j = id - o * K2;
    float v;
    if (j < 2 * HID_DIM) {
        const int h = j >> 1;
        v = (j & 1) ? -C_im[o * HID_DIM + h] : C_re[o * HID_DIM + h];
    } else {
        v = D[o * IN_DIM + (j - 2 * HID_DIM)];
    }
    W2[id] = (__bf16)v;
}

// ---------------------------------------------------------------------------
// Convert X (f32) into the X-columns of Z (bf16, row stride K2)
// ---------------------------------------------------------------------------
__global__ void lru_convert_x(const float* __restrict__ X, __bf16* __restrict__ Z)
{
    const int id = blockIdx.x * blockDim.x + threadIdx.x;
    if (id >= MROWS * IN_DIM) return;
    const int m = id >> 8;
    const int i = id & (IN_DIM - 1);
    Z[(size_t)m * K2 + 2 * HID_DIM + i] = (__bf16)X[id];
}

// ---------------------------------------------------------------------------
// Blocked scan, phase 1: per (channel, segment) local scan with h0=0;
// store segment-end state to carries[s][c].  262144 threads.
// ---------------------------------------------------------------------------
__global__ void lru_scan_local(const float* __restrict__ drive,
                               const float* __restrict__ lam,
                               float2* __restrict__ carries)
{
    const int id = blockIdx.x * blockDim.x + threadIdx.x;
    if (id >= NCHAN * SEGS) return;
    const int c = id & (NCHAN - 1);
    const int s = id >> 13;              // / NCHAN
    const int b = c >> 9;
    const int h = c & (HID_DIM - 1);
    const float2 l = *(const float2*)&lam[2 * h];
    float hr = 0.f, hi = 0.f;
    size_t row = (size_t)b * TSTEPS + (size_t)s * SEGLEN;
    for (int t = 0; t < SEGLEN; ++t, ++row) {
        const float2 d = *(const float2*)&drive[row * N1 + 2 * h];
        const float nr = l.x * hr - l.y * hi + d.x;
        const float ni = l.x * hi + l.y * hr + d.y;
        hr = nr; hi = ni;
    }
    carries[(size_t)s * NCHAN + c] = make_float2(hr, hi);
}

// ---------------------------------------------------------------------------
// Phase 2: sequential combine across 32 segments per channel (8192 threads):
// rewrite carries[s][c] with the carry-IN of segment s.
// ---------------------------------------------------------------------------
__global__ void lru_scan_combine(const float* __restrict__ lamL,
                                 float2* __restrict__ carries)
{
    const int c = blockIdx.x * blockDim.x + threadIdx.x;
    if (c >= NCHAN) return;
    const int h = c & (HID_DIM - 1);
    const float2 L = *(const float2*)&lamL[2 * h];
    float cr = 0.f, ci = 0.f;
    for (int s = 0; s < SEGS; ++s) {
        const float2 loc = carries[(size_t)s * NCHAN + c];
        carries[(size_t)s * NCHAN + c] = make_float2(cr, ci);
        const float nr = L.x * cr - L.y * ci + loc.x;
        const float ni = L.x * ci + L.y * cr + loc.y;
        cr = nr; ci = ni;
    }
}

// ---------------------------------------------------------------------------
// Phase 3: re-run each segment from its carry-in; write H into Z (packed bf16).
// ---------------------------------------------------------------------------
__global__ void lru_scan_apply(const float* __restrict__ drive,
                               const float* __restrict__ lam,
                               const float2* __restrict__ carries,
                               __bf16* __restrict__ Z)
{
    const int id = blockIdx.x * blockDim.x + threadIdx.x;
    if (id >= NCHAN * SEGS) return;
    const int c = id & (NCHAN - 1);
    const int s = id >> 13;
    const int b = c >> 9;
    const int h = c & (HID_DIM - 1);
    const float2 l  = *(const float2*)&lam[2 * h];
    const float2 h0 = carries[(size_t)s * NCHAN + c];
    float hr = h0.x, hi = h0.y;
    size_t row = (size_t)b * TSTEPS + (size_t)s * SEGLEN;
    for (int t = 0; t < SEGLEN; ++t, ++row) {
        const float2 d = *(const float2*)&drive[row * N1 + 2 * h];
        const float nr = l.x * hr - l.y * hi + d.x;
        const float ni = l.x * hi + l.y * hr + d.y;
        hr = nr; hi = ni;
        v2bf pk = { (__bf16)hr, (__bf16)hi };
        *(v2bf*)&Z[row * K2 + 2 * h] = pk;     // one 4B store (re,im)
    }
}

// ---------------------------------------------------------------------------
// Launch
// ---------------------------------------------------------------------------
extern "C" void kernel_launch(void* const* d_in, const int* in_sizes, int n_in,
                              void* d_out, int out_size, void* d_ws, size_t ws_size,
                              hipStream_t stream)
{
    const float* X         = (const float*)d_in[0];
    const float* nu_log    = (const float*)d_in[1];
    const float* theta_log = (const float*)d_in[2];
    const float* B_re      = (const float*)d_in[3];
    const float* B_im      = (const float*)d_in[4];
    const float* C_re      = (const float*)d_in[5];
    const float* C_im      = (const float*)d_in[6];
    const float* D         = (const float*)d_in[7];
    const float* bh_re     = (const float*)d_in[8];
    const float* bh_im     = (const float*)d_in[9];
    const float* bias_out  = (const float*)d_in[10];
    float* Y               = (float*)d_out;

    // workspace carve-up (256B aligned blocks)
    char* ws = (char*)d_ws;
    size_t off = 0;
    auto carve = [&](size_t bytes) { void* p = ws + off; off += (bytes + 255) & ~(size_t)255; return p; };
    __bf16* W1      = (__bf16*)carve((size_t)N1 * K1 * sizeof(__bf16));     //  0.5 MB
    __bf16* W2      = (__bf16*)carve((size_t)N2 * K2 * sizeof(__bf16));     //  0.6 MB
    float*  bias1   = (float*) carve((size_t)N1 * sizeof(float));
    float*  lam     = (float*) carve((size_t)2 * HID_DIM * sizeof(float));
    float*  lamL    = (float*) carve((size_t)2 * HID_DIM * sizeof(float));
    float2* carries = (float2*)carve((size_t)SEGS * NCHAN * sizeof(float2)); //  2 MB
    float*  drive   = (float*) carve((size_t)MROWS * N1 * sizeof(float));   //  268 MB
    __bf16* Z       = (__bf16*)carve((size_t)MROWS * K2 * sizeof(__bf16));  //  168 MB

    // 1. pack weights (+ lambda, lambda^SEGLEN)
    lru_pack_w1<<<(N1 + 255) / 256, 256, 0, stream>>>(nu_log, theta_log, B_re, B_im,
                                                      bh_re, bh_im, W1, bias1, lam, lamL);
    lru_pack_w2<<<(N2 * K2 + 255) / 256, 256, 0, stream>>>(C_re, C_im, D, W2);
    // 2. X -> Z (bf16)
    lru_convert_x<<<(MROWS * IN_DIM) / 256, 256, 0, stream>>>(X, Z);
    // 3. GEMM1: drive = Xbf16(Z cols 1024.., lda=K2) * W1^T + bias1
    {
        dim3 grid(N1 / BN, MROWS / BM);
        lru_gemm_bf16<<<grid, 256, 0, stream>>>(Z + 2 * HID_DIM, K2,
                                                W1, bias1, drive, MROWS, N1, K1);
    }
    // 4. blocked scan -> H columns of Z (bf16)
    lru_scan_local  <<<(NCHAN * SEGS) / 256, 256, 0, stream>>>(drive, lam, carries);
    lru_scan_combine<<<NCHAN / 256,          256, 0, stream>>>(lamL, carries);
    lru_scan_apply  <<<(NCHAN * SEGS) / 256, 256, 0, stream>>>(drive, lam, carries, Z);
    // 5. GEMM2: Y = Z * W2^T + bias_out
    {
        dim3 grid(N2 / BN, MROWS / BM);
        lru_gemm_bf16<<<grid, 256, 0, stream>>>(Z, K2, W2, bias_out, Y, MROWS, N2, K2);
    }
}
